// RCSLS_28922309771771
// MI455X (gfx1250) — compile-verified, matrix operand order checked by reading, not run on hardware
//
#include <hip/hip_runtime.h>
#include <hip/hip_bf16.h>

typedef __attribute__((ext_vector_type(8)))  float  v8f;
typedef __attribute__((ext_vector_type(16))) __bf16 v16bf;

union FragBF { v16bf v; uint4 q[2]; };

#define DIMD   512
#define NROWS  4096
#define MROWS  50000
#define MPAD   50176       // 28 partitions * 56 strips * 32 cols
#define CSPLIT 28          // column partitions per GEMM
#define SPB    56          // 32-col strips per partition block (7 per wave)
#define NEGBIG (-3.0e38f)

// ---------------------------------------------------------------------------
// Kernel 1: row-wise L2 normalize fp32 -> bf16  (one block per row, 256 thr)
//           rows >= realRows are zero-filled padding.
// ---------------------------------------------------------------------------
__global__ __launch_bounds__(256) void norm_to_bf16(const float* __restrict__ src,
                                                    __bf16* __restrict__ dst,
                                                    int realRows) {
    const int row = blockIdx.x;
    const int tid = threadIdx.x;
    if (row >= realRows) {
        ((unsigned int*)(dst + (size_t)row * DIMD))[tid] = 0u;
        return;
    }
    const float2 v = ((const float2*)(src + (size_t)row * DIMD))[tid];
    float ss = v.x * v.x + v.y * v.y;
    for (int o = 16; o > 0; o >>= 1) ss += __shfl_down(ss, o, 32);
    __shared__ float part[8];
    if ((tid & 31) == 0) part[tid >> 5] = ss;
    __syncthreads();
    float tot = part[0] + part[1] + part[2] + part[3] +
                part[4] + part[5] + part[6] + part[7];
    const float s = rsqrtf(tot);
    union { __bf16 h[2]; unsigned int u; } pk;
    pk.h[0] = (__bf16)(v.x * s);
    pk.h[1] = (__bf16)(v.y * s);
    ((unsigned int*)(dst + (size_t)row * DIMD))[tid] = pk.u;
}

// ---------------------------------------------------------------------------
// Kernel 2: frow[r] = 2 * dot(x,y) / (|x||y|)   (one block per row)
// ---------------------------------------------------------------------------
__global__ __launch_bounds__(256) void fterm(const float* __restrict__ X,
                                             const float* __restrict__ Y,
                                             float* __restrict__ frow) {
    const int row = blockIdx.x;
    const int tid = threadIdx.x;
    const float2 x = ((const float2*)(X + (size_t)row * DIMD))[tid];
    const float2 y = ((const float2*)(Y + (size_t)row * DIMD))[tid];
    float sx = x.x * x.x + x.y * x.y;
    float sy = y.x * y.x + y.y * y.y;
    float dd = x.x * y.x + x.y * y.y;
    for (int o = 16; o > 0; o >>= 1) {
        sx += __shfl_down(sx, o, 32);
        sy += __shfl_down(sy, o, 32);
        dd += __shfl_down(dd, o, 32);
    }
    __shared__ float part[8][3];
    if ((tid & 31) == 0) {
        part[tid >> 5][0] = sx; part[tid >> 5][1] = sy; part[tid >> 5][2] = dd;
    }
    __syncthreads();
    if (tid == 0) {
        float tsx = 0.f, tsy = 0.f, tdd = 0.f;
        for (int i = 0; i < 8; ++i) { tsx += part[i][0]; tsy += part[i][1]; tdd += part[i][2]; }
        frow[row] = 2.0f * tdd * rsqrtf(tsx) * rsqrtf(tsy);
    }
}

// ---------------------------------------------------------------------------
// Kernel 3: fused bf16-WMMA GEMM + streaming top-10
//   grid = (NROWS/32, CSPLIT), block = 256 (8 waves)
//   Block tile: 32 rows (A staged in LDS; compiler hoists A frags to regs).
//   Wave tile 32x32: 2 A-frags x 2 B-frags -> 4 WMMAs per k-step, double-
//   buffered so loads of step kk+1 overlap the WMMAs of step kk.
//   Merge phase: LDS pool overlays the A tile; 4 chunks of 8 rows,
//   hierarchical scan (8 thr/row -> 1 thr/row), 3 barriers per chunk.
//   cand layout: [row][CSPLIT][10]
// ---------------------------------------------------------------------------
__global__ __launch_bounds__(256) void gemm_topk(const __bf16* __restrict__ A,
                                                 const __bf16* __restrict__ B,
                                                 float* __restrict__ cand) {
    // 42.5 KB: [0,32KB) A tile during GEMM; after a barrier, overlaid by
    // pool[8][1280] (40KB) + mini[8][8][10] (2.5KB) for the merge.
    __shared__ __align__(16) unsigned char smemRaw[(8 * 1280 + 8 * 8 * 10) * 4];

    const int rowTile = blockIdx.x;       // 0..127
    const int part    = blockIdx.y;       // 0..27
    const int tid     = threadIdx.x;
    const int wave    = tid >> 5;
    const int lane    = tid & 31;
    const int lane16  = lane & 15;
    const int khalf   = (lane >> 4) * 8;   // K sub-offset for A/B fragments

    __bf16* As = (__bf16*)smemRaw;

    // ---- stage A tile (already normalized bf16) into LDS: 2048 uint4 ----
    {
        const uint4* asrc = (const uint4*)(A + (size_t)rowTile * 32 * DIMD);
        uint4* adst = (uint4*)smemRaw;
        for (int i = tid; i < 32 * DIMD / 8; i += 256) adst[i] = asrc[i];
    }
    __syncthreads();

    // ---- per-lane top-10 lists (16 rows/lane, disjoint column streams) ----
    float topk[16][10];
#pragma unroll
    for (int r = 0; r < 16; ++r)
#pragma unroll
        for (int j = 0; j < 10; ++j) topk[r][j] = NEGBIG;

    const __bf16* arow0 = As + lane16 * DIMD + khalf;          // rows 0..15
    const __bf16* arow1 = As + (16 + lane16) * DIMD + khalf;   // rows 16..31

    // ---- main loop: each wave owns 7 of the block's 56 32-col strips ----
    for (int s = wave; s < SPB; s += 8) {
        const int colBase = (part * SPB + s) * 32;
        const __bf16* brow0 = B + (size_t)(colBase + lane16) * DIMD + khalf;
        const __bf16* brow1 = brow0 + (size_t)16 * DIMD;

        FragBF a[2][2];          // [buffer][row-group]
        FragBF b[2][2];          // [buffer][col-group]
        v8f c[2][2] = {{{}, {}}, {{}, {}}};   // [row-group][col-group]

        auto loadFrags = [&](int bi, int kk) {
            const int kb = kk * 32;
            a[bi][0].q[0] = *(const uint4*)(arow0 + kb);        // ds_load_b128
            a[bi][0].q[1] = *(const uint4*)(arow0 + kb + 16);
            a[bi][1].q[0] = *(const uint4*)(arow1 + kb);
            a[bi][1].q[1] = *(const uint4*)(arow1 + kb + 16);
            b[bi][0].q[0] = *(const uint4*)(brow0 + kb);        // global_load_b128
            b[bi][0].q[1] = *(const uint4*)(brow0 + kb + 16);
            b[bi][1].q[0] = *(const uint4*)(brow1 + kb);
            b[bi][1].q[1] = *(const uint4*)(brow1 + kb + 16);
        };

        loadFrags(0, 0);
#pragma unroll
        for (int kk = 0; kk < 16; ++kk) {
            const int cur = kk & 1;
            if (kk < 15) loadFrags(cur ^ 1, kk + 1);   // prefetch next k-step
#pragma unroll
            for (int rg = 0; rg < 2; ++rg)
#pragma unroll
                for (int cg = 0; cg < 2; ++cg)
                    c[rg][cg] = __builtin_amdgcn_wmma_f32_16x16x32_bf16(
                                    false, a[cur][rg].v, false, b[cur][cg].v,
                                    (short)0, c[rg][cg], false, false);
        }

        // streaming top-10 update (fully unrolled, static indices -> registers)
#pragma unroll
        for (int cg = 0; cg < 2; ++cg) {
            if (colBase + cg * 16 + lane16 < MROWS) {   // exclude padded columns
#pragma unroll
                for (int rg = 0; rg < 2; ++rg) {
#pragma unroll
                    for (int g = 0; g < 8; ++g) {
                        const float v = c[rg][cg][g];
                        float (&lst)[10] = topk[rg * 8 + g];
                        if (v > lst[9]) {
                            lst[9] = v;
#pragma unroll
                            for (int q = 9; q > 0; --q) {
                                if (lst[q] > lst[q - 1]) {
                                    float tmp = lst[q - 1];
                                    lst[q - 1] = lst[q];
                                    lst[q]     = tmp;
                                }
                            }
                        }
                    }
                }
            }
        }
    }
    __syncthreads();   // all strips done; safe to overlay As

    // ---- merge: 4 chunks of 8 rows; 3 barriers per chunk ----
    // chunk -> rows (chunk>>1)*16 + (chunk&1)*8 + g, owned by lane half
    // (lane>>4)==(chunk&1), topk bank (chunk>>1)*8 + g.
    float* pool = (float*)smemRaw;                 // [8 rows][1280]
    float* mini = ((float*)smemRaw) + 8 * 1280;    // [8 rows][8][10]

#pragma unroll
    for (int chunk = 0; chunk < 4; ++chunk) {
        if ((lane >> 4) == (chunk & 1)) {
#pragma unroll
            for (int g = 0; g < 8; ++g)
#pragma unroll
                for (int j = 0; j < 10; ++j)
                    pool[g * 1280 + (wave * 16 + lane16) * 10 + j] =
                        topk[(chunk >> 1) * 8 + g][j];
        }
        __syncthreads();

        // stage 1: 8 threads per row, each scans 160 candidates
        if (tid < 64) {
            const int g   = tid >> 3;
            const int sub = tid & 7;
            float top[10];
#pragma unroll
            for (int j = 0; j < 10; ++j) top[j] = NEGBIG;
            const float* src = pool + g * 1280 + sub * 160;
            for (int i = 0; i < 160; ++i) {
                const float v = src[i];
                if (v > top[9]) {
                    top[9] = v;
#pragma unroll
                    for (int q = 9; q > 0; --q)
                        if (top[q] > top[q - 1]) {
                            float t = top[q - 1]; top[q - 1] = top[q]; top[q] = t;
                        }
                }
            }
#pragma unroll
            for (int j = 0; j < 10; ++j) mini[(g * 8 + sub) * 10 + j] = top[j];
        }
        __syncthreads();

        // stage 2: 1 thread per row merges 80 -> sorted top-10 -> global
        if (tid < 8) {
            const int g = tid;
            float top[10];
#pragma unroll
            for (int j = 0; j < 10; ++j) top[j] = NEGBIG;
            const float* src = mini + g * 80;
            for (int i = 0; i < 80; ++i) {
                const float v = src[i];
                if (v > top[9]) {
                    top[9] = v;
#pragma unroll
                    for (int q = 9; q > 0; --q)
                        if (top[q] > top[q - 1]) {
                            float t = top[q - 1]; top[q - 1] = top[q]; top[q] = t;
                        }
                }
            }
            const int row = rowTile * 32 + (chunk >> 1) * 16 + (chunk & 1) * 8 + g;
#pragma unroll
            for (int j = 0; j < 10; ++j)
                cand[((size_t)row * CSPLIT + part) * 10 + j] = top[j];
        }
        __syncthreads();
    }
}

// ---------------------------------------------------------------------------
// Kernel 4: merge CSPLIT partitions per (matrix,row); sum first knn values
// ---------------------------------------------------------------------------
__global__ __launch_bounds__(256) void finalize_topk(const float* __restrict__ cand,
                                                     float* __restrict__ fkrow,
                                                     const int* __restrict__ knnPtr) {
    const int idx = blockIdx.x * blockDim.x + threadIdx.x;   // [0, 2*NROWS)
    if (idx >= 2 * NROWS) return;
    const float* c = cand + (size_t)idx * (CSPLIT * 10);
    float top[10];
#pragma unroll
    for (int j = 0; j < 10; ++j) top[j] = NEGBIG;
    for (int i = 0; i < CSPLIT * 10; ++i) {
        const float v = c[i];
        if (v > top[9]) {
            top[9] = v;
#pragma unroll
            for (int j = 9; j > 0; --j) {
                if (top[j] > top[j - 1]) {
                    float tmp = top[j - 1]; top[j - 1] = top[j]; top[j] = tmp;
                }
            }
        }
    }
    const int knn = *knnPtr;
    float s = 0.f;
#pragma unroll
    for (int j = 0; j < 10; ++j)
        if (j < knn) s += top[j];
    fkrow[idx] = s;
}

// ---------------------------------------------------------------------------
// Kernel 5: deterministic final reduction -> scalar loss
// ---------------------------------------------------------------------------
__global__ __launch_bounds__(256) void final_reduce(const float* __restrict__ frow,
                                                    const float* __restrict__ fkrow,
                                                    const int* __restrict__ knnPtr,
                                                    float* __restrict__ out) {
    __shared__ float sm[256];
    const int tid = threadIdx.x;
    float a = 0.f, b = 0.f, cc = 0.f;
    for (int i = tid; i < NROWS; i += 256) {
        a  += frow[i];
        b  += fkrow[i];
        cc += fkrow[NROWS + i];
    }
    float f, fk0, fk1;
    sm[tid] = a;  __syncthreads();
    for (int s = 128; s > 0; s >>= 1) { if (tid < s) sm[tid] += sm[tid + s]; __syncthreads(); }
    f = sm[0];    __syncthreads();
    sm[tid] = b;  __syncthreads();
    for (int s = 128; s > 0; s >>= 1) { if (tid < s) sm[tid] += sm[tid + s]; __syncthreads(); }
    fk0 = sm[0];  __syncthreads();
    sm[tid] = cc; __syncthreads();
    for (int s = 128; s > 0; s >>= 1) { if (tid < s) sm[tid] += sm[tid + s]; __syncthreads(); }
    fk1 = sm[0];
    if (tid == 0) {
        const float k = (float)(*knnPtr);
        const float F = f - fk0 / k - fk1 / k;
        out[0] = -F / (float)NROWS;
    }
}

// ---------------------------------------------------------------------------
extern "C" void kernel_launch(void* const* d_in, const int* in_sizes, int n_in,
                              void* d_out, int out_size, void* d_ws, size_t ws_size,
                              hipStream_t stream) {
    const float* X  = (const float*)d_in[0];   // X_trans  [N,512]
    const float* Y  = (const float*)d_in[1];   // Y_tgt    [N,512]
    const float* Zs = (const float*)d_in[2];   // Z_src    [M,512]
    const float* Zt = (const float*)d_in[3];   // Z_tgt    [M,512]
    const float* Bk = (const float*)d_in[4];   // back_emb [N,512]
    const int*  knn = (const int*)d_in[5];     // scalar

    char* ws = (char*)d_ws;
    size_t off = 0;
    auto carve = [&](size_t bytes) -> void* {
        void* p = ws + off;
        off = (off + bytes + 255) & ~(size_t)255;
        return p;
    };
    __bf16* ZtBF = (__bf16*)carve((size_t)MPAD * DIMD * 2);    // 51.4 MB
    __bf16* ZsBF = (__bf16*)carve((size_t)MPAD * DIMD * 2);    // 51.4 MB
    __bf16* XnBF = (__bf16*)carve((size_t)NROWS * DIMD * 2);   //  4.2 MB
    __bf16* BkBF = (__bf16*)carve((size_t)NROWS * DIMD * 2);   //  4.2 MB
    float*  frow  = (float*)carve((size_t)NROWS * 4);
    float*  cand  = (float*)carve((size_t)2 * NROWS * CSPLIT * 10 * 4); // 9.2 MB
    float*  fkrow = (float*)carve((size_t)2 * NROWS * 4);

    // Phase 1: normalization + bf16 conversion (bandwidth-trivial)
    norm_to_bf16<<<MPAD, 256, 0, stream>>>(Zt, ZtBF, MROWS);
    norm_to_bf16<<<MPAD, 256, 0, stream>>>(Zs, ZsBF, MROWS);
    norm_to_bf16<<<NROWS, 256, 0, stream>>>(X,  XnBF, NROWS);
    norm_to_bf16<<<NROWS, 256, 0, stream>>>(Bk, BkBF, NROWS);
    fterm<<<NROWS, 256, 0, stream>>>(X, Y, frow);

    // Phase 2: the two compute-bound WMMA GEMMs with fused streaming top-10
    dim3 grid(NROWS / 32, CSPLIT);
    gemm_topk<<<grid, 256, 0, stream>>>(XnBF, ZtBF, cand);
    gemm_topk<<<grid, 256, 0, stream>>>(BkBF, ZsBF, cand + (size_t)NROWS * CSPLIT * 10);

    // Phase 3: merges + scalar
    finalize_topk<<<(2 * NROWS + 255) / 256, 256, 0, stream>>>(cand, fkrow, knn);
    final_reduce<<<1, 256, 0, stream>>>(frow, fkrow, knn, (float*)d_out);
}
